// SageSparseLinearAttentionImpl_35442070126842
// MI455X (gfx1250) — matrix operand order repaired and checked by reading, not compiled, not run on hardware
//
#include <hip/hip_runtime.h>
#include <cstdint>
#include <cstddef>

// ---------------------------------------------------------------------------
// Problem constants (reference: B=1, L=4096, H=8, D=128, BLKQ=128, BLKK=64,
// TOPK_RATIO=0.5 -> topk=32, EPS=1e-5)
// ---------------------------------------------------------------------------
static constexpr int Hh   = 8;
static constexpr int Ll   = 4096;
static constexpr int Dd   = 128;
static constexpr int NQ   = 32;    // L / BLKQ
static constexpr int NK   = 64;    // L / BLKK
static constexpr int TOPK = 32;
static constexpr int BQ   = 128;
static constexpr int BK   = 64;

typedef __bf16 bf16_t;
typedef __attribute__((ext_vector_type(16))) __bf16 v16bf;
typedef __attribute__((ext_vector_type(8)))  __bf16 v8bf;
typedef __attribute__((ext_vector_type(8)))  float  v8f;

#define DEV static __device__ __forceinline__

DEV bf16_t f2bf(float f) { return (bf16_t)f; }   // native cvt

DEV v8f vzero() {
  v8f z;
  #pragma unroll
  for (int i = 0; i < 8; ++i) z[i] = 0.0f;
  return z;
}

// load 16 contiguous bf16 (32B, 16B-aligned) as one fragment
DEV v16bf ld16(const bf16_t* p) {
  v8bf lo = *(const v8bf*)p;
  v8bf hi = *(const v8bf*)(p + 8);
  return __builtin_shufflevector(lo, hi, 0,1,2,3,4,5,6,7,8,9,10,11,12,13,14,15);
}
// join two 8-element runs (each 16B-aligned) into a fragment
DEV v16bf join8(const bf16_t* p0, const bf16_t* p1) {
  v8bf lo = *(const v8bf*)p0;
  v8bf hi = *(const v8bf*)p1;
  return __builtin_shufflevector(lo, hi, 0,1,2,3,4,5,6,7,8,9,10,11,12,13,14,15);
}

DEV v8f wmma_bf16(v16bf a, v16bf b, v8f c) {
  // D = A(16x32 bf16) * B(32x16 bf16) + C(f32)
  return __builtin_amdgcn_wmma_f32_16x16x32_bf16(
      /*neg_a=*/false, a, /*neg_b=*/false, b,
      /*c_mod=*/(short)0, c, /*reuse_a=*/false, /*reuse_b=*/false);
}

// reductions within a 16-lane half-wave (C-layout rows live per 16-lane group)
DEV float redmax16(float x) {
  #pragma unroll
  for (int mm = 1; mm < 16; mm <<= 1) x = fmaxf(x, __shfl_xor(x, mm, 32));
  return x;
}
DEV float redsum16(float x) {
  #pragma unroll
  for (int mm = 1; mm < 16; mm <<= 1) x += __shfl_xor(x, mm, 32);
  return x;
}
DEV float redsum32(float x) {
  #pragma unroll
  for (int mm = 1; mm < 32; mm <<= 1) x += __shfl_xor(x, mm, 32);
  return x;
}

// async global -> LDS copy of 16 bytes (per lane); tracked by ASYNCcnt
DEV void async_cp16(unsigned lds_addr, const void* gaddr) {
  asm volatile("global_load_async_to_lds_b128 %0, %1, off"
               :: "v"(lds_addr), "v"(gaddr) : "memory");
}
DEV void async_wait0() {
  asm volatile("s_wait_asynccnt 0x0" ::: "memory");
}

// ---------------------------------------------------------------------------
// 1. kmean[h][d] = mean_l k[l,h,d]
// ---------------------------------------------------------------------------
__global__ void kmean_kernel(const float* __restrict__ k, float* __restrict__ kmean) {
  int hx = blockIdx.x, d = threadIdx.x;
  float s = 0.f;
  for (int l = 0; l < Ll; ++l) s += k[((size_t)l * Hh + hx) * Dd + d];
  kmean[hx * Dd + d] = s * (1.0f / Ll);
}

// ---------------------------------------------------------------------------
// 2. pq[h][qb][d] = mean over 128 rows ; pk[h][kb][d] = mean over 64 - kmean
// ---------------------------------------------------------------------------
__global__ void pq_kernel(const float* __restrict__ q, float* __restrict__ pq) {
  int qb = blockIdx.x, hx = blockIdx.y, d = threadIdx.x;
  float s = 0.f;
  for (int r = 0; r < BQ; ++r) s += q[((size_t)(qb * BQ + r) * Hh + hx) * Dd + d];
  pq[(hx * NQ + qb) * Dd + d] = s * (1.0f / BQ);
}
__global__ void pk_kernel(const float* __restrict__ k, const float* __restrict__ kmean,
                          float* __restrict__ pk) {
  int kb = blockIdx.x, hx = blockIdx.y, d = threadIdx.x;
  float s = 0.f;
  for (int r = 0; r < BK; ++r) s += k[((size_t)(kb * BK + r) * Hh + hx) * Dd + d];
  pk[(hx * NK + kb) * Dd + d] = s * (1.0f / BK) - kmean[hx * Dd + d];
}

// ---------------------------------------------------------------------------
// 3. score[qb][kb] = pq . pk ; rank-select top-32 (ties: lower index first,
//    matching jax.lax.top_k ordering)
// ---------------------------------------------------------------------------
__global__ void topk_kernel(const float* __restrict__ pq, const float* __restrict__ pk,
                            int* __restrict__ lut) {
  int qb = blockIdx.x, hx = blockIdx.y, kb = threadIdx.x;   // 64 threads
  __shared__ float spq[Dd];
  __shared__ float sc[NK];
  for (int d = kb; d < Dd; d += 64) spq[d] = pq[(hx * NQ + qb) * Dd + d];
  __syncthreads();
  const float* pkr = pk + (size_t)(hx * NK + kb) * Dd;
  float s = 0.f;
  for (int d = 0; d < Dd; ++d) s += spq[d] * pkr[d];
  sc[kb] = s;
  __syncthreads();
  float mys = sc[kb];
  int rank = 0;
  for (int j = 0; j < NK; ++j) {
    float o = sc[j];
    rank += (o > mys) || (o == mys && j < kb);
  }
  if (rank < TOPK) lut[(hx * NQ + qb) * TOPK + rank] = kb;
}

// ---------------------------------------------------------------------------
// 4. fq / fk : softmax over D per row (one wave per row)
// ---------------------------------------------------------------------------
DEV void softmax_row(const float* __restrict__ src, float* __restrict__ dst, int lane) {
  float x[4];
  float mx = -1e30f;
  #pragma unroll
  for (int j = 0; j < 4; ++j) { x[j] = src[lane + 32 * j]; mx = fmaxf(mx, x[j]); }
  #pragma unroll
  for (int mm = 1; mm < 32; mm <<= 1) mx = fmaxf(mx, __shfl_xor(mx, mm, 32));
  float s = 0.f;
  #pragma unroll
  for (int j = 0; j < 4; ++j) { x[j] = __expf(x[j] - mx); s += x[j]; }
  s = redsum32(s);
  float inv = 1.0f / s;
  #pragma unroll
  for (int j = 0; j < 4; ++j) dst[lane + 32 * j] = x[j] * inv;
}

__global__ __launch_bounds__(256) void rowsoftmax_kernel(
    const float* __restrict__ q, const float* __restrict__ k,
    float* __restrict__ fq, float* __restrict__ fk) {
  int wave = threadIdx.x >> 5, lane = threadIdx.x & 31;
  size_t row = (size_t)blockIdx.x * 8 + wave;   // row over (l*H + h)
  softmax_row(q + row * Dd, fq + row * Dd, lane);
  softmax_row(k + row * Dd, fk + row * Dd, lane);
}

__global__ void zero_kernel(float* __restrict__ p, int n) {
  int i = blockIdx.x * 256 + threadIdx.x;
  if (i < n) p[i] = 0.0f;
}

// ---------------------------------------------------------------------------
// 5a. kbf[h][l][d] = bf16(k[l,h,d])  (head-major bf16 copy of K)
// ---------------------------------------------------------------------------
__global__ void kbf_kernel(const float* __restrict__ k, bf16_t* __restrict__ kbf) {
  int l = blockIdx.x, hx = blockIdx.y, d = threadIdx.x;
  kbf[((size_t)hx * Ll + l) * Dd + d] = f2bf(k[((size_t)l * Hh + hx) * Dd + d]);
}

// ---------------------------------------------------------------------------
// 5b. vbt[h][kb][d][key] = bf16(v[kb*64+key, h, d])  (per-block transposed V)
//     Transpose done once here (through LDS) instead of 32x in the hot loop.
// ---------------------------------------------------------------------------
__global__ __launch_bounds__(256) void vbt_kernel(const float* __restrict__ v,
                                                  bf16_t* __restrict__ vbt) {
  int kb = blockIdx.x, hx = blockIdx.y;
  int tid = threadIdx.x;
  __shared__ alignas(16) bf16_t sVt[Dd][72];
  int srow = tid >> 2, sseg = (tid & 3) * 32;
  const float* vp = v + ((size_t)(kb * BK + srow) * Hh + hx) * Dd + sseg;
  #pragma unroll
  for (int j = 0; j < 32; ++j) sVt[sseg + j][srow] = f2bf(vp[j]);
  __syncthreads();
  bf16_t* dst = vbt + ((size_t)hx * NK + kb) * Dd * BK;   // [d][key] tile
  int d = tid >> 1, key = (tid & 1) * 32;
  #pragma unroll
  for (int j2 = 0; j2 < 4; ++j2)
    *(v8bf*)(dst + (size_t)d * BK + key + j2 * 8) = *(const v8bf*)&sVt[d][key + j2 * 8];
}

// ---------------------------------------------------------------------------
// 6. ksum[h][d] = sum_l fk[l,h,d]
// ---------------------------------------------------------------------------
__global__ void ksum_kernel(const float* __restrict__ fk, float* __restrict__ ksum) {
  int hx = blockIdx.x, d = threadIdx.x;
  float s = 0.f;
  for (int l = 0; l < Ll; ++l) s += fk[((size_t)l * Hh + hx) * Dd + d];
  ksum[hx * Dd + d] = s;
}

// ---------------------------------------------------------------------------
// 7. kv[h][d][e] = sum_l fk[l,h,d] * v[l,h,e] via WMMA. LDS tiles stored
//    TRANSPOSED ([d][l], [e][l]) so A and B fragments are contiguous loads.
//    Grid (H, 4 L-chunks), accumulate with f32 atomics.
// ---------------------------------------------------------------------------
__global__ __launch_bounds__(256) void kv_kernel(
    const float* __restrict__ fk, const float* __restrict__ v, float* __restrict__ kv) {
  int hx = blockIdx.x, chunk = blockIdx.y;
  int tid = threadIdx.x, wave = tid >> 5, lane = tid & 31;
  int m = lane & 15, hf = lane >> 4;
  __shared__ alignas(16) bf16_t sFt[Dd][40];   // [d][l] transposed, stride 40
  __shared__ alignas(16) bf16_t sVt[Dd][40];   // [e][l] transposed
  v8f acc[8];
  #pragma unroll
  for (int t = 0; t < 8; ++t) acc[t] = vzero();

  int r = tid >> 3, seg = (tid & 7) * 16;       // staging: 8 threads per l-row
  for (int li = 0; li < 32; ++li) {
    int l0 = chunk * 1024 + li * 32;
    __syncthreads();
    {
      const float* fp = fk + ((size_t)(l0 + r) * Hh + hx) * Dd + seg;
      const float* vp = v  + ((size_t)(l0 + r) * Hh + hx) * Dd + seg;
      #pragma unroll
      for (int j = 0; j < 16; ++j) { sFt[seg + j][r] = f2bf(fp[j]); sVt[seg + j][r] = f2bf(vp[j]); }
    }
    __syncthreads();
    // A fragment: M = d (wave slice), K = 32 l's: two contiguous 8-runs
    int dcol = wave * 16 + m;
    v16bf a = join8(&sFt[dcol][hf * 8], &sFt[dcol][16 + hf * 8]);
    // batch all 8 B fragments, then issue the WMMAs (partial dscnt waits)
    v16bf bfr[8];
    #pragma unroll
    for (int nt = 0; nt < 8; ++nt) bfr[nt] = join8(&sVt[nt * 16 + m][hf * 8], &sVt[nt * 16 + m][16 + hf * 8]);
    #pragma unroll
    for (int nt = 0; nt < 8; ++nt) acc[nt] = wmma_bf16(a, bfr[nt], acc[nt]);
  }
  #pragma unroll
  for (int nt = 0; nt < 8; ++nt) {
    #pragma unroll
    for (int g = 0; g < 8; ++g) {
      int d = wave * 16 + g + 8 * hf;
      int e = nt * 16 + m;
      atomicAdd(&kv[((size_t)hx * Dd + d) * Dd + e], acc[nt][g]);
    }
  }
}

// ---------------------------------------------------------------------------
// 8. Block-sparse flash attention. Grid (NQ, H), 256 threads = 8 waves.
//    Double-buffered LDS staging with GLOBAL_LOAD_ASYNC_TO_LDS_B128:
//    next block's K/V tiles stream into the alternate buffer while the
//    current one is consumed by WMMA (latency hidden by one compute phase).
// ---------------------------------------------------------------------------
__global__ __launch_bounds__(256) void sparse_attn_kernel(
    const float* __restrict__ q, const bf16_t* __restrict__ kbf,
    const bf16_t* __restrict__ vbt, const int* __restrict__ lut,
    float* __restrict__ out) {
  int qb = blockIdx.x, hx = blockIdx.y;
  int tid = threadIdx.x, wave = tid >> 5, lane = tid & 31;
  int m = lane & 15, hf = lane >> 4;

  __shared__ alignas(16) bf16_t sKb[2][BK][136];     // [key][d] row-major padded
  __shared__ alignas(16) bf16_t sVtb[2][Dd][72];     // [d][key] transposed padded
  __shared__ alignas(16) bf16_t sPb[8][16][72];      // per-wave P scratch

  const float scale = 0.0883883476483184f;   // 1/sqrt(128)

  // Q A-fragments (4 chunks of K=32 over D=128), rows = qb*128 + wave*16 + m
  int qrow = qb * BQ + wave * 16 + m;
  const float* qr = q + ((size_t)qrow * Hh + hx) * Dd;
  v16bf qa[4];
  #pragma unroll
  for (int kc = 0; kc < 4; ++kc) {
    int d0 = kc * 32 + hf * 8;
    #pragma unroll
    for (int e = 0; e < 8; ++e) qa[kc][e]     = f2bf(qr[d0 + e]);
    #pragma unroll
    for (int e = 0; e < 8; ++e) qa[kc][8 + e] = f2bf(qr[d0 + 16 + e]);
  }

  v8f o[8];
  #pragma unroll
  for (int t = 0; t < 8; ++t) o[t] = vzero();
  float mrow[8], lrow[8];
  #pragma unroll
  for (int g = 0; g < 8; ++g) { mrow[g] = -1e30f; lrow[g] = 0.f; }

  int lbase = (hx * NQ + qb) * TOPK;
  int srow = tid >> 2, sseg = (tid & 3) * 32;   // K staging: 4 threads/row, 32 d
  int vd = tid >> 1, vkey = (tid & 1) * 32;     // V staging: 2 threads/d, 32 keys
  const bf16_t* kbase = kbf + (size_t)hx * Ll * Dd;
  const bf16_t* vbase = vbt + (size_t)hx * NK * Dd * BK;

  // issue async K/V tile copies for block kb into LDS buffer `buf`
  auto stage_async = [&](int kb, int buf) {
    const bf16_t* kp = kbase + ((size_t)kb * BK + srow) * Dd + sseg;
    const bf16_t* vp = vbase + (size_t)kb * Dd * BK + (size_t)vd * BK + vkey;
    unsigned kl = (unsigned)(size_t)&sKb[buf][srow][sseg];
    unsigned vl = (unsigned)(size_t)&sVtb[buf][vd][vkey];
    #pragma unroll
    for (int j2 = 0; j2 < 4; ++j2) {
      async_cp16(kl + j2 * 16, kp + j2 * 8);
      async_cp16(vl + j2 * 16, vp + j2 * 8);
    }
  };

  stage_async(lut[lbase], 0);

  for (int t = 0; t < TOPK; ++t) {
    int buf = t & 1;
    async_wait0();        // this wave's loads for `buf` are in LDS
    __syncthreads();      // all waves' loads done; all done reading buf^1
    if (t + 1 < TOPK) stage_async(lut[lbase + t + 1], buf ^ 1);

    const bf16_t (*sK)[136] = sKb[buf];
    const bf16_t (*sVt)[72] = sVtb[buf];

    // S = Q * K^T : 4 column tiles (64 keys), 4 K-chunks each (batched loads)
    v8f s[4];
    #pragma unroll
    for (int nt = 0; nt < 4; ++nt) {
      v16bf bb[4];
      #pragma unroll
      for (int kc = 0; kc < 4; ++kc)
        bb[kc] = join8(&sK[nt * 16 + m][kc * 32 + hf * 8], &sK[nt * 16 + m][kc * 32 + 16 + hf * 8]);
      v8f acc = vzero();
      #pragma unroll
      for (int kc = 0; kc < 4; ++kc) acc = wmma_bf16(qa[kc], bb[kc], acc);
      s[nt] = acc;
    }

    // online softmax per row (row = g + 8*hf; stats reduced across 16-lane group)
    #pragma unroll
    for (int g = 0; g < 8; ++g) {
      float mx = -1e30f;
      #pragma unroll
      for (int nt = 0; nt < 4; ++nt) { s[nt][g] *= scale; mx = fmaxf(mx, s[nt][g]); }
      mx = redmax16(mx);
      float mnew = fmaxf(mrow[g], mx);
      float corr = __expf(mrow[g] - mnew);
      mrow[g] = mnew;
      lrow[g] *= corr;
      #pragma unroll
      for (int ot = 0; ot < 8; ++ot) o[ot][g] *= corr;
      float ps = 0.f;
      #pragma unroll
      for (int nt = 0; nt < 4; ++nt) {
        float p = __expf(s[nt][g] - mnew);
        s[nt][g] = p;
        ps += p;
      }
      lrow[g] += redsum16(ps);
    }

    // P (C layout) -> wave-private LDS (bf16) -> A fragments (K = 64 keys, 2 chunks)
    bf16_t* pw = &sPb[wave][0][0];
    #pragma unroll
    for (int nt = 0; nt < 4; ++nt) {
      #pragma unroll
      for (int g = 0; g < 8; ++g)
        pw[(g + 8 * hf) * 72 + nt * 16 + m] = f2bf(s[nt][g]);
    }

    v16bf pa[2];
    const bf16_t* pr = pw + m * 72;
    #pragma unroll
    for (int kc = 0; kc < 2; ++kc) {
      int k0 = kc * 32 + hf * 8;
      pa[kc] = join8(pr + k0, pr + k0 + 16);
    }

    // O += P * V : 8 output column tiles, 2 key chunks (batched loads)
    #pragma unroll
    for (int ot = 0; ot < 8; ++ot) {
      v16bf vb0 = join8(&sVt[ot * 16 + m][hf * 8], &sVt[ot * 16 + m][16 + hf * 8]);
      v16bf vb1 = join8(&sVt[ot * 16 + m][32 + hf * 8], &sVt[ot * 16 + m][48 + hf * 8]);
      o[ot] = wmma_bf16(pa[0], vb0, o[ot]);
      o[ot] = wmma_bf16(pa[1], vb1, o[ot]);
    }
  }

  // epilogue: O /= l , write o_s to out [l,h,d]
  #pragma unroll
  for (int ot = 0; ot < 8; ++ot) {
    #pragma unroll
    for (int g = 0; g < 8; ++g) {
      int r = qb * BQ + wave * 16 + g + 8 * hf;
      int d = ot * 16 + m;
      out[((size_t)r * Hh + hx) * Dd + d] = o[ot][g] / lrow[g];
    }
  }
}

// ---------------------------------------------------------------------------
// 9. Linear attention epilogue: num = fq*kv (WMMA), den = eps + fq.ksum,
//    o_l = num/den, proj = o_l @ w^T (WMMA), out += proj + b.
//    kv[h] is staged once per WG as a TRANSPOSED bf16 LDS tile [e][d].
// ---------------------------------------------------------------------------
__global__ __launch_bounds__(256) void linear_out_kernel(
    const float* __restrict__ fq, const float* __restrict__ kv,
    const float* __restrict__ ksum, const float* __restrict__ w,
    const float* __restrict__ b, float* __restrict__ out) {
  int blk = blockIdx.x, hx = blockIdx.y;
  int tid = threadIdx.x, wave = tid >> 5, lane = tid & 31;
  int m = lane & 15, hf = lane >> 4;
  __shared__ float sKs[Dd];
  __shared__ float sB[Dd];
  __shared__ alignas(16) bf16_t kvT[Dd][136];    // [e][d] transposed, padded
  __shared__ alignas(16) bf16_t sPb[8][16][72];
  if (tid < Dd) { sKs[tid] = ksum[hx * Dd + tid]; sB[tid] = b[tid]; }
  // stage kv[h] transposed as bf16 (coalesced global reads over e)
  #pragma unroll 4
  for (int j = 0; j < 64; ++j) {
    int flat = j * 256 + tid;
    int d = flat >> 7, e = flat & 127;
    kvT[e][d] = f2bf(kv[((size_t)hx * Dd + d) * Dd + e]);
  }
  __syncthreads();

  int row = blk * 128 + wave * 16 + m;
  const float* fqr = fq + ((size_t)row * Hh + hx) * Dd;
  v16bf a[4];
  float den = 0.f;
  #pragma unroll
  for (int kc = 0; kc < 4; ++kc) {
    int d0 = kc * 32 + hf * 8;
    #pragma unroll
    for (int e = 0; e < 8; ++e) { float x = fqr[d0 + e];      a[kc][e]     = f2bf(x); den += x * sKs[d0 + e]; }
    #pragma unroll
    for (int e = 0; e < 8; ++e) { float x = fqr[d0 + 16 + e]; a[kc][8 + e] = f2bf(x); den += x * sKs[d0 + 16 + e]; }
  }
  den += __shfl_xor(den, 16, 32);   // lanes m and m+16 now both hold den[row m]
  den += 1e-5f;

  // num tiles; divide by den; store o_l (bf16) in wave-private LDS
  bf16_t* pw = &sPb[wave][0][0];
  #pragma unroll
  for (int nt = 0; nt < 8; ++nt) {
    v16bf bb[4];
    #pragma unroll
    for (int kc = 0; kc < 4; ++kc)
      bb[kc] = join8(&kvT[nt * 16 + m][kc * 32 + hf * 8], &kvT[nt * 16 + m][kc * 32 + 16 + hf * 8]);
    v8f acc = vzero();
    #pragma unroll
    for (int kc = 0; kc < 4; ++kc) acc = wmma_bf16(a[kc], bb[kc], acc);
    #pragma unroll
    for (int g = 0; g < 8; ++g) {
      float dg = __shfl(den, g + 8 * hf, 32);  // den of row g+8*hf
      pw[(g + 8 * hf) * 72 + nt * 16 + m] = f2bf(acc[g] / dg);
    }
  }

  // o_l A fragments (K = 128 over 4 chunks)
  v16bf pa[4];
  const bf16_t* pr = pw + m * 72;
  #pragma unroll
  for (int kc = 0; kc < 4; ++kc) {
    int d0 = kc * 32 + hf * 8;
    pa[kc] = join8(pr + d0, pr + d0 + 16);
  }

  // projection: out[e2] = sum_e o_l[e] * w[e2][e]  (w rows are contiguous)
  #pragma unroll
  for (int nt = 0; nt < 8; ++nt) {
    v16bf wb[4];
    int e2 = nt * 16 + m;
    #pragma unroll
    for (int kc = 0; kc < 4; ++kc) {
      const float* wr = w + (size_t)e2 * Dd + kc * 32 + hf * 8;
      #pragma unroll
      for (int ei = 0; ei < 16; ++ei) wb[kc][ei] = f2bf(wr[ei + (ei >= 8 ? 8 : 0)]);
    }
    v8f acc = vzero();
    #pragma unroll
    for (int kc = 0; kc < 4; ++kc) acc = wmma_bf16(pa[kc], wb[kc], acc);
    #pragma unroll
    for (int g = 0; g < 8; ++g) {
      int r = blk * 128 + wave * 16 + g + 8 * hf;
      size_t oi = ((size_t)r * Hh + hx) * Dd + e2;
      out[oi] += acc[g] + sB[e2];
    }
  }
}

// ---------------------------------------------------------------------------
// launch
// ---------------------------------------------------------------------------
extern "C" void kernel_launch(void* const* d_in, const int* in_sizes, int n_in,
                              void* d_out, int out_size, void* d_ws, size_t ws_size,
                              hipStream_t stream) {
  (void)in_sizes; (void)n_in; (void)out_size; (void)ws_size;
  const float* q = (const float*)d_in[0];   // [L,H,D]
  const float* k = (const float*)d_in[1];
  const float* v = (const float*)d_in[2];
  const float* w = (const float*)d_in[3];   // [D,D] row-major: w[e_out][d_in]
  const float* b = (const float*)d_in[4];   // [D]
  float* out = (float*)d_out;               // [L,H,D]

  // workspace layout (floats): ~51 MB total
  float*  ws    = (float*)d_ws;
  float*  kmean = ws;                          // 1024
  float*  pq    = kmean + 1024;                // 32768
  float*  pk    = pq + 32768;                  // 65536
  int*    lut   = (int*)(pk + 65536);          // 8192 ints
  float*  fq    = (float*)(lut + 8192);        // 4194304
  float*  fk    = fq + (size_t)Ll * Hh * Dd;   // 4194304
  float*  kv    = fk + (size_t)Ll * Hh * Dd;   // 131072
  float*  ksum  = kv + Hh * Dd * Dd;           // 1024
  bf16_t* kbf   = (bf16_t*)(ksum + 1024);      // H*L*D bf16  (8.4 MB)
  bf16_t* vbt   = kbf + (size_t)Hh * Ll * Dd;  // H*NK*D*BK bf16 (8.4 MB)

  kmean_kernel<<<dim3(Hh), dim3(Dd), 0, stream>>>(k, kmean);
  pq_kernel<<<dim3(NQ, Hh), dim3(Dd), 0, stream>>>(q, pq);
  pk_kernel<<<dim3(NK, Hh), dim3(Dd), 0, stream>>>(k, kmean, pk);
  topk_kernel<<<dim3(NQ, Hh), dim3(64), 0, stream>>>(pq, pk, lut);
  kbf_kernel<<<dim3(Ll, Hh), dim3(Dd), 0, stream>>>(k, kbf);
  vbt_kernel<<<dim3(NK, Hh), dim3(256), 0, stream>>>(v, vbt);
  rowsoftmax_kernel<<<dim3(Ll * Hh / 8), dim3(256), 0, stream>>>(q, k, fq, fk);
  zero_kernel<<<dim3((Hh * Dd * Dd + 255) / 256), dim3(256), 0, stream>>>(kv, Hh * Dd * Dd);
  kv_kernel<<<dim3(Hh, 4), dim3(256), 0, stream>>>(fk, v, kv);
  ksum_kernel<<<dim3(Hh), dim3(Dd), 0, stream>>>(fk, ksum);
  sparse_attn_kernel<<<dim3(NQ, Hh), dim3(256), 0, stream>>>(q, kbf, vbt, lut, out);
  linear_out_kernel<<<dim3(Ll / BQ, Hh), dim3(256), 0, stream>>>(fq, kv, ksum, w, b, out);
}
